// Conv2d_1589137899949
// MI455X (gfx1250) — matrix operand order repaired and checked by reading, not compiled, not run on hardware
//
#include <hip/hip_runtime.h>

// ---------------------------------------------------------------------------
// Fused conv_downsample_2d for MI455X (gfx1250):
//   FIR(4x4, pad 2,2) + conv3x3 stride2  ==  single 6x6 stride-2 conv with
//   weff = w (x) k2d.  Implicit GEMM on v_wmma_f32_16x16x32_bf16.
//
// v2: 4 waves/WG, each wave owns a 32x32 output macro-tile (2 oh rows x 16 ow
//     x 32 co) -> 4 WMMA per K-chunk from 8 ds_load_b128 (ds/wmma = 2.0,
//     was 3.0) and 4 independent accumulation chains per wave.
// ---------------------------------------------------------------------------

typedef __attribute__((ext_vector_type(16))) __bf16 v16bf;
typedef __attribute__((ext_vector_type(8)))  __bf16 v8bf;
typedef __attribute__((ext_vector_type(8)))  float  v8f;
typedef __attribute__((ext_vector_type(4)))  float  v4f;

#define NB   8
#define CI   32
#define CO   32
#define HH   768
#define WW   768
#define OHE  384
#define OWE  384

#define OH_T 8      // output rows per workgroup (2 per wave, 4 waves)
#define OW_T 16     // output cols per workgroup (M dim of WMMA)

#define XT_H 20     // input tile rows  = 2*OH_T + 6 - 2
#define XT_W 36     // input tile cols  = 2*OW_T + 6 - 2
#define CELL 80     // bytes per cell: 32 ci * 2B bf16 = 64B data + 16B pad

#define XS_BYTES   (XT_H * XT_W * CELL)          // 57600
#define WS_BYTES   (36 * CO * CELL)              // 92160
#define SMEM_BYTES (XS_BYTES + WS_BYTES)         // 149760 (2 WGs / 320KB WGP)

#define WP_ELEMS   (36 * CO * CI)                // 36864 bf16 in workspace
#define NTHREADS   128                           // 4 waves

// ---------------------------------------------------------------------------
// Prep: effective 6x6 weights, bf16, layout [rs=6*r+s][co][ci] (K-major for
// the WMMA chunk ordering k = rs*32 + ci).
// ---------------------------------------------------------------------------
__global__ void prep_weights_kernel(const float* __restrict__ w,
                                    __bf16* __restrict__ wp) {
    int idx = blockIdx.x * blockDim.x + threadIdx.x;
    if (idx >= WP_ELEMS) return;
    int ci = idx & 31;
    int co = (idx >> 5) & 31;
    int rs = idx >> 10;
    int r = rs / 6, s = rs - 6 * (rs / 6);

    const float kv[4] = {1.f, 3.f, 3.f, 1.f};   // outer(kv,kv)/64 == FIR k2d
    float acc = 0.f;
    #pragma unroll
    for (int kh = 0; kh < 3; ++kh) {
        int fh = r - kh;
        if (fh < 0 || fh > 3) continue;
        #pragma unroll
        for (int kw = 0; kw < 3; ++kw) {
            int fw = s - kw;
            if (fw < 0 || fw > 3) continue;
            acc += w[((co * CI + ci) * 3 + kh) * 3 + kw] *
                   (kv[fh] * kv[fw] * (1.0f / 64.0f));
        }
    }
    wp[idx] = (__bf16)acc;
}

// ---------------------------------------------------------------------------
static __device__ __forceinline__ v16bf cat8(v8bf lo, v8bf hi) {
    return __builtin_shufflevector(lo, hi, 0, 1, 2, 3, 4, 5, 6, 7,
                                           8, 9, 10, 11, 12, 13, 14, 15);
}

static __device__ __forceinline__ v16bf load_frag(const char* cell, int half) {
    v8bf lo = *(const v8bf*)(cell + half * 16);
    v8bf hi = *(const v8bf*)(cell + 32 + half * 16);
    return cat8(lo, hi);
}

static __device__ __forceinline__ void store_acc(float* o, v8f c, float bz) {
    v4f lo = {c[0] + bz, c[1] + bz, c[2] + bz, c[3] + bz};
    v4f hi = {c[4] + bz, c[5] + bz, c[6] + bz, c[7] + bz};
    *(v4f*)o       = lo;
    *(v4f*)(o + 4) = hi;
}

// ---------------------------------------------------------------------------
// Main fused kernel. Grid: (OW/16, OH/8, N). Block: 128 threads = 4 waves.
// Wave w computes output rows oh0+2w, oh0+2w+1, 16 ow values, all 32 co.
// ---------------------------------------------------------------------------
__global__ __launch_bounds__(NTHREADS, 1)
void fused_downconv_kernel(const float* __restrict__ x,
                           const __bf16* __restrict__ wp,
                           const float* __restrict__ bias,
                           float* __restrict__ out) {
    extern __shared__ char smem[];
    char* xs  = smem;                // [XT_H][XT_W] cells of CELL bytes (ci-major)
    char* wsm = smem + XS_BYTES;     // [36][CO]     cells of CELL bytes (ci-major)

    const int tid = threadIdx.x;
    const int n   = blockIdx.z;
    const int oh0 = blockIdx.y * OH_T;
    const int ow0 = blockIdx.x * OW_T;

    // ---- Stage weights: global bf16 [36][32][32] -> padded LDS cells ------
    // 4608 uint4 groups (16B = 8 bf16), 36 per thread, exact.
    {
        const uint4* __restrict__ wp4 = (const uint4*)wp;
        for (int g = tid; g < (WP_ELEMS * 2) / 16; g += NTHREADS) {
            int cellidx = g >> 2;            // (rs*32 + co)
            int sub     = (g & 3) * 16;      // 8-bf16 sub-chunk within cell
            *(uint4*)(wsm + cellidx * CELL + sub) = wp4[g];
        }
    }

    // ---- Stage input halo tile: fp32 global -> bf16 padded LDS ------------
    // 32ci * 20h * 36w = 23040 elems, 180 per thread, exact; zero-pad edges.
    {
        const int h_base = 2 * oh0 - 2;
        const int w_base = 2 * ow0 - 2;
        for (int i = tid; i < CI * XT_H * XT_W; i += NTHREADS) {
            int ci  = i / (XT_H * XT_W);
            int rem = i - ci * (XT_H * XT_W);
            int hl  = rem / XT_W;
            int wl  = rem - hl * XT_W;
            int hg  = h_base + hl;
            int wg  = w_base + wl;
            float v = 0.f;
            if (hg >= 0 && hg < HH && wg >= 0 && wg < WW)
                v = x[((n * CI + ci) * HH + hg) * (long)WW + wg];
            *(__bf16*)(xs + (hl * XT_W + wl) * CELL + ci * 2) = (__bf16)v;
        }
    }
    __syncthreads();

    // ---- Compute: K = 36 chunks of (one 6x6 tap) x (32 ci) ----------------
    const int lane = tid & 31;
    const int wave = tid >> 5;           // 0..3 -> oh rows 2w, 2w+1
    const int half = lane >> 4;          // lane group (ISA K-split)
    const int mrow = lane & 15;          // A row / D column index

    v8f c00 = {};                        // rows 2w   , co  0..15
    v8f c01 = {};                        // rows 2w   , co 16..31
    v8f c10 = {};                        // rows 2w+1 , co  0..15
    v8f c11 = {};                        // rows 2w+1 , co 16..31

    #pragma unroll
    for (int r = 0; r < 6; ++r) {
        const char* xrow0 = xs + (4 * wave + r) * (XT_W * CELL);
        const char* xrow1 = xrow0 + 2 * (XT_W * CELL);
        #pragma unroll
        for (int s = 0; s < 6; ++s) {
            const int rs = r * 6 + s;
            // A fragments for the two oh rows (ISA 16-bit A layout: lane
            // half selects ci groups {8h..8h+7} and {16+8h..23+8h}).
            const int coloff = (2 * mrow + s) * CELL;
            v16bf a0 = load_frag(xrow0 + coloff, half);
            v16bf a1 = load_frag(xrow1 + coloff, half);

            // B fragments: col n = co; lane half selects K(ci) {8h..8h+7} and {16+8h..23+8h}.
            const char* bc0 = wsm + (rs * CO + mrow) * CELL + half * 16;
            v8bf b0lo = *(const v8bf*)(bc0);
            v8bf b0hi = *(const v8bf*)(bc0 + 32);
            v16bf b0  = cat8(b0lo, b0hi);
            const char* bc1 = bc0 + 16 * CELL;
            v8bf b1lo = *(const v8bf*)(bc1);
            v8bf b1hi = *(const v8bf*)(bc1 + 32);
            v16bf b1  = cat8(b1lo, b1hi);

            c00 = __builtin_amdgcn_wmma_f32_16x16x32_bf16(
                      false, a0, false, b0, (short)0, c00, false, false);
            c01 = __builtin_amdgcn_wmma_f32_16x16x32_bf16(
                      false, a0, false, b1, (short)0, c01, false, false);
            c10 = __builtin_amdgcn_wmma_f32_16x16x32_bf16(
                      false, a1, false, b0, (short)0, c10, false, false);
            c11 = __builtin_amdgcn_wmma_f32_16x16x32_bf16(
                      false, a1, false, b1, (short)0, c11, false, false);
        }
    }

    // ---- Epilogue: bias + store. D layout: lane = col(co), VGPR v = row ---
    // v+8*half  ->  each lane holds 8 consecutive ow -> two b128 stores/acc.
    const int oh   = oh0 + 2 * wave;
    const int owb  = ow0 + half * 8;
    const float b0z = bias[mrow];
    const float b1z = bias[16 + mrow];

    float* o00 = out + (((long)(n * CO + mrow)      * OHE + oh)     * OWE + owb);
    float* o01 = out + (((long)(n * CO + 16 + mrow) * OHE + oh)     * OWE + owb);
    float* o10 = out + (((long)(n * CO + mrow)      * OHE + oh + 1) * OWE + owb);
    float* o11 = out + (((long)(n * CO + 16 + mrow) * OHE + oh + 1) * OWE + owb);

    store_acc(o00, c00, b0z);
    store_acc(o01, c01, b1z);
    store_acc(o10, c10, b0z);
    store_acc(o11, c11, b1z);
}

// ---------------------------------------------------------------------------
extern "C" void kernel_launch(void* const* d_in, const int* in_sizes, int n_in,
                              void* d_out, int out_size, void* d_ws, size_t ws_size,
                              hipStream_t stream) {
    (void)in_sizes; (void)n_in; (void)out_size; (void)ws_size;

    const float* x = (const float*)d_in[0];   // [8,32,768,768] f32
    const float* w = (const float*)d_in[1];   // [32,32,3,3]    f32
    const float* b = (const float*)d_in[2];   // [32]           f32
    float* out     = (float*)d_out;           // [8,32,384,384] f32
    __bf16* wp     = (__bf16*)d_ws;           // 36*32*32 bf16 effective weights

    prep_weights_kernel<<<(WP_ELEMS + 255) / 256, 256, 0, stream>>>(w, wp);

    dim3 grid(OWE / OW_T, OHE / OH_T, NB);    // (24, 48, 8)
    fused_downconv_kernel<<<grid, NTHREADS, SMEM_BYTES, stream>>>(x, wp, b, out);
}
